// NeuronAttentionBase_31318901522471
// MI455X (gfx1250) — compile-verified, matrix operand order checked by reading, not run on hardware
//
#include <hip/hip_runtime.h>
#include <hip/hip_bf16.h>
#include <stdint.h>

// ---------------------------------------------------------------------------
// Llama-style GQA attention block for MI455X (gfx1250, wave32, WMMA).
// All matmuls use V_WMMA_F32_16X16X32_BF16 (bf16 in, f32 accumulate).
// GEMM A-tiles staged via the Tensor Data Mover (6-arg clang-23 builtin).
// ---------------------------------------------------------------------------

#if __has_builtin(__builtin_amdgcn_tensor_load_to_lds) && \
    __has_builtin(__builtin_amdgcn_s_wait_tensorcnt)
#define USE_TDM 1
#else
#define USE_TDM 0
#endif

namespace {

constexpr int Bc   = 2;
constexpr int Sc   = 2048;
constexpr int HIDc = 4096;
constexpr int NHc  = 32;
constexpr int NKVc = 8;
constexpr int Dc   = 128;
constexpr int GROUPSc = NHc / NKVc;
constexpr int Mrows = Bc * Sc;   // 4096 token rows

typedef __attribute__((ext_vector_type(16))) __bf16   v16bf;
typedef __attribute__((ext_vector_type(8)))  float    v8f;
typedef __attribute__((ext_vector_type(4)))  uint32_t v4u;

#if USE_TDM
typedef __attribute__((ext_vector_type(4))) unsigned int tdm_g0_t;  // D# group0
typedef __attribute__((ext_vector_type(8))) int          tdm_g1_t;  // D# group1
typedef __attribute__((ext_vector_type(4))) int          tdm_g2_t;  // D# group2/3
#endif

union FragU { v16bf v; v4u q[2]; };
static_assert(sizeof(FragU) == 32, "fragment must be 8 VGPRs");

__device__ __forceinline__ uint16_t f2bf(float f) {
  uint32_t x = __float_as_uint(f);
  return (uint16_t)((x + 0x7FFFu + ((x >> 16) & 1u)) >> 16);  // RNE
}
__device__ __forceinline__ float bf2f(uint16_t h) {
  return __uint_as_float(((uint32_t)h) << 16);
}

// Load a 16(row) x 32(K) bf16 operand fragment in the ISA 7.12.2 layout:
// lanes L: row = r0 + L%16, hi = L/16; K-pairs: i<4 -> hi*4+i, i>=4 -> 8+hi*4+(i-4).
// tile_u = uint32 (bf16-pair) view, row_u = row pitch in uint32 units (mult of 4).
__device__ __forceinline__ v16bf load_frag(const uint32_t* tile_u, int r0, int row_u) {
  const int lane = (int)(threadIdx.x & 31u);
  const int r  = r0 + (lane & 15);
  const int hi = lane >> 4;
  const uint32_t* base = tile_u + (size_t)r * (size_t)row_u + hi * 4;
  FragU f;
  f.q[0] = *reinterpret_cast<const v4u*>(base);      // K pairs (hi*4 .. hi*4+3)
  f.q[1] = *reinterpret_cast<const v4u*>(base + 8);  // K pairs (8+hi*4 .. )
  return f.v;
}

__device__ __forceinline__ v8f wmma_bf16(v16bf a, v16bf b, v8f c) {
  return __builtin_amdgcn_wmma_f32_16x16x32_bf16(false, a, false, b, (short)0, c,
                                                 false, false);
}

// ---------------------------------------------------------------------------
// fp32 -> bf16 cast
// ---------------------------------------------------------------------------
__global__ __launch_bounds__(512) void cast_f32_bf16(const float* __restrict__ src,
                                                     uint16_t* __restrict__ dst,
                                                     size_t n) {
  size_t i = (size_t)blockIdx.x * blockDim.x + threadIdx.x;
  if (i < n) dst[i] = f2bf(src[i]);
}

// ---------------------------------------------------------------------------
// C[M,N] (f32 or bf16) = A[M,K] @ B[K,N], A/B bf16 row-major.
// Block: 128 thr (4 waves), tile 128x128, BK=32. Each wave: 64x64 patch
// (4x4 WMMA accumulators -> 16 WMMAs per 8 fragment loads per K-step).
// A tile staged by the Tensor Data Mover (wave 0 issues the descriptor,
// s_wait_tensorcnt + barrier publishes it); B tile staged transposed in VALU.
// ---------------------------------------------------------------------------
template <bool OUT_BF16>
__global__ __launch_bounds__(128) void gemm_bf16_kernel(
    const uint16_t* __restrict__ A, const uint16_t* __restrict__ B,
    void* __restrict__ C, int N, int K) {
  __shared__ uint32_t As[128 * 16];   // [row][16 K-pairs]  (8 KB)
  __shared__ uint32_t BsU[128 * 16];  // B^T tile: [n][16 K-pairs]
  uint16_t* Bs = reinterpret_cast<uint16_t*>(BsU);

  const int m0 = blockIdx.y * 128;
  const int n0 = blockIdx.x * 128;
  const int tid = (int)threadIdx.x;
  const int lane = tid & 31;
  const int wid = tid >> 5;
  const int wm = (wid & 1) * 64;   // wave row offset inside tile
  const int wn = (wid >> 1) * 64;  // wave col offset inside tile

  v8f acc[4][4];
#pragma unroll
  for (int i = 0; i < 4; ++i)
#pragma unroll
    for (int j = 0; j < 4; ++j)
#pragma unroll
      for (int e = 0; e < 8; ++e) acc[i][j][e] = 0.0f;

  const uint32_t* Au = reinterpret_cast<const uint32_t*>(A);
  const uint32_t* Bu = reinterpret_cast<const uint32_t*>(B);
  const int Ku = K >> 1;
  const int Nu = N >> 1;

#if USE_TDM
  const uint32_t lds_a = (uint32_t)(uintptr_t)(&As[0]);      // LDS byte offset
  const uint64_t ga_row0 = (uint64_t)(uintptr_t)A + (uint64_t)m0 * (uint64_t)K * 2u;
#endif

  for (int k0 = 0; k0 < K; k0 += 32) {
    __syncthreads();

#if USE_TDM
    if (wid == 0) {
      // 2D TDM descriptor (ISA 8.3/8.4): tile 32(K, contiguous) x 128(rows),
      // data_size=2B, row pitch = K elements, destination = As (row-major).
      const uint64_t gaddr = ga_row0 + (uint64_t)k0 * 2u;
      tdm_g0_t g0;
      g0[0] = 1u;                                   // count=1, user descriptor
      g0[1] = lds_a;                                // lds_addr
      g0[2] = (uint32_t)gaddr;                      // global_addr[31:0]
      g0[3] = (uint32_t)((gaddr >> 32) & 0x1FFFFFFu) | (2u << 30);  // addr hi | type=2
      tdm_g1_t g1;
      g1[0] = (int)(1u << 16);                      // wg_mask=0, data_size=1 (2B)
      g1[1] = (int)(((uint32_t)K & 0xFFFFu) << 16);            // tensor_dim0 lo
      g1[2] = (int)(((uint32_t)K >> 16) |
                    (((uint32_t)Mrows & 0xFFFFu) << 16));      // dim0 hi | dim1 lo
      g1[3] = (int)(((uint32_t)Mrows >> 16) | (32u << 16));    // dim1 hi | tile_dim0=32
      g1[4] = (int)128u;                            // tile_dim1=128, tile_dim2=0
      g1[5] = (int)(uint32_t)K;                     // tensor_dim0_stride[31:0]
      g1[6] = 0;                                    // stride hi | dim1_stride lo
      g1[7] = 0;
      tdm_g2_t z4 = {0, 0, 0, 0};
      tdm_g1_t z8 = {0, 0, 0, 0, 0, 0, 0, 0};
      __builtin_amdgcn_tensor_load_to_lds(g0, g1, z4, z4, z8, 0);
    }
#else
    // Manual A staging: 128x32 bf16 = 2048 uints, fragment-friendly row-major.
#pragma unroll
    for (int i = 0; i < 16; ++i) {
      int lin = tid + 128 * i;
      int r = lin >> 4, c = lin & 15;
      As[lin] = Au[(size_t)(m0 + r) * Ku + (k0 >> 1) + c];
    }
#endif

    // B tile 32x128, stored transposed ([n][k]) so fragments are contiguous.
#pragma unroll
    for (int i = 0; i < 16; ++i) {
      int lin = tid + 128 * i;
      int kr = lin >> 6, np = lin & 63;
      uint32_t w = Bu[(size_t)(k0 + kr) * Nu + (n0 >> 1) + np];
      Bs[(2 * np) * 32 + kr]     = (uint16_t)(w & 0xFFFFu);
      Bs[(2 * np + 1) * 32 + kr] = (uint16_t)(w >> 16);
    }

#if USE_TDM
    if (wid == 0) __builtin_amdgcn_s_wait_tensorcnt(0);
#endif
    __syncthreads();

    v16bf af[4], bfr[4];
#pragma unroll
    for (int i = 0; i < 4; ++i) af[i] = load_frag(As, wm + i * 16, 16);
#pragma unroll
    for (int j = 0; j < 4; ++j) bfr[j] = load_frag(BsU, wn + j * 16, 16);
#pragma unroll
    for (int i = 0; i < 4; ++i)
#pragma unroll
      for (int j = 0; j < 4; ++j) acc[i][j] = wmma_bf16(af[i], bfr[j], acc[i][j]);
  }

  const int hi = lane >> 4, ln = lane & 15;
#pragma unroll
  for (int i = 0; i < 4; ++i)
#pragma unroll
    for (int j = 0; j < 4; ++j)
#pragma unroll
      for (int v = 0; v < 8; ++v) {
        size_t m = (size_t)(m0 + wm + i * 16 + v + 8 * hi);
        size_t n = (size_t)(n0 + wn + j * 16 + ln);
        float val = acc[i][j][v];
        if (OUT_BF16)
          reinterpret_cast<uint16_t*>(C)[m * (size_t)N + n] = f2bf(val);
        else
          reinterpret_cast<float*>(C)[m * (size_t)N + n] = val;
      }
}

// ---------------------------------------------------------------------------
// RoPE + relayout: P [B*S, heads*D] bf16 -> Out [B, heads, S, D] bf16
// ---------------------------------------------------------------------------
__global__ __launch_bounds__(256) void rope_relayout(
    const uint16_t* __restrict__ P, uint16_t* __restrict__ Out,
    const float* __restrict__ cosc, const float* __restrict__ sinc,
    const int* __restrict__ pos, int heads) {
  size_t idx = (size_t)blockIdx.x * blockDim.x + threadIdx.x;
  size_t total = (size_t)Bc * Sc * heads * (Dc / 2);
  if (idx >= total) return;
  int d = (int)(idx % (Dc / 2)); size_t t = idx / (Dc / 2);
  int hh = (int)(t % heads); t /= heads;
  int s = (int)(t % Sc); int b = (int)(t / Sc);
  size_t prow = ((size_t)(b * Sc + s) * heads + hh) * Dc;
  float q1 = bf2f(P[prow + d]);
  float q2 = bf2f(P[prow + d + Dc / 2]);
  int p = pos[b * Sc + s];
  float c1 = cosc[(size_t)p * Dc + d],          s1 = sinc[(size_t)p * Dc + d];
  float c2 = cosc[(size_t)p * Dc + d + Dc / 2], s2 = sinc[(size_t)p * Dc + d + Dc / 2];
  size_t orow = (((size_t)b * heads + hh) * Sc + s) * Dc;
  Out[orow + d]          = f2bf(q1 * c1 - q2 * s1);   // rotate_half: -x2 part
  Out[orow + d + Dc / 2] = f2bf(q2 * c2 + q1 * s2);   // +x1 part
}

// V: [B*S, NKV*D] -> Vt [B, NKV, D, S]  (so PV's B^T view is contiguous in t)
__global__ __launch_bounds__(256) void v_transpose(const uint16_t* __restrict__ Vp,
                                                   uint16_t* __restrict__ Vt) {
  size_t idx = (size_t)blockIdx.x * blockDim.x + threadIdx.x;
  size_t total = (size_t)Bc * Sc * NKVc * Dc;
  if (idx >= total) return;
  int d = (int)(idx % Dc); size_t t = idx / Dc;
  int hk = (int)(t % NKVc); t /= NKVc;
  int s = (int)(t % Sc); int b = (int)(t / Sc);
  uint16_t v = Vp[((size_t)(b * Sc + s) * NKVc + hk) * Dc + d];
  Vt[(((size_t)b * NKVc + hk) * Dc + d) * Sc + s] = v;
}

// ---------------------------------------------------------------------------
// Flash attention: 1 wave / block, 16 Q rows, KV blocks of 32, causal.
// Qa [B,NH,S,D], Ka [B,NKV,S,D], Vt [B,NKV,D,S] (all bf16).
// Writes attn as bf16 in [B*S, NH*D] layout (o_proj GEMM input).
// ---------------------------------------------------------------------------
__global__ __launch_bounds__(32) void attn_kernel(
    const uint16_t* __restrict__ Qa, const uint16_t* __restrict__ Ka,
    const uint16_t* __restrict__ Vt, uint16_t* __restrict__ Ob) {
  __shared__ uint32_t PtU[16 * 16];                 // 16x32 bf16 P tile
  uint16_t* Pt = reinterpret_cast<uint16_t*>(PtU);

  const int lane = (int)(threadIdx.x & 31u);
  const int hi = lane >> 4, ln = lane & 15;
  const int q0 = blockIdx.x * 16;
  const int h  = blockIdx.y;
  const int b  = blockIdx.z;
  const int hk = h / GROUPSc;

  const uint32_t* Qh = reinterpret_cast<const uint32_t*>(Qa) +
                       (size_t)(b * NHc + h) * Sc * (Dc / 2);
  const uint32_t* Kh = reinterpret_cast<const uint32_t*>(Ka) +
                       (size_t)(b * NKVc + hk) * Sc * (Dc / 2);
  const uint32_t* Vh = reinterpret_cast<const uint32_t*>(Vt) +
                       (size_t)(b * NKVc + hk) * Dc * (Sc / 2);

  v16bf qf[4];
#pragma unroll
  for (int ks = 0; ks < 4; ++ks) qf[ks] = load_frag(Qh + ks * 16, q0, Dc / 2);

  v8f o[8];
#pragma unroll
  for (int nb = 0; nb < 8; ++nb)
#pragma unroll
    for (int e = 0; e < 8; ++e) o[nb][e] = 0.0f;
  float mprev[8], lsum[8];
#pragma unroll
  for (int v = 0; v < 8; ++v) { mprev[v] = -3.0e38f; lsum[v] = 0.0f; }

  const float rscale = 0.088388347648318447f;  // 1/sqrt(128)
  const int jmax = (q0 + 15) >> 5;             // inclusive KV block index

  for (int j = 0; j <= jmax; ++j) {
    const int t0 = j * 32;

    // Prefetch next KV block's K rows (one 256B row per lane) -> global_prefetch.
    if (j < jmax)
      __builtin_prefetch(Kh + (size_t)(t0 + 32 + lane) * (Dc / 2), 0, 0);

    v8f s[2];
#pragma unroll
    for (int nt = 0; nt < 2; ++nt)
#pragma unroll
      for (int e = 0; e < 8; ++e) s[nt][e] = 0.0f;
    // scores = Q K^T : B^T of K^T is row-major K -> direct global fragments
#pragma unroll
    for (int ks = 0; ks < 4; ++ks) {
      v16bf kf0 = load_frag(Kh + ks * 16, t0, Dc / 2);
      v16bf kf1 = load_frag(Kh + ks * 16, t0 + 16, Dc / 2);
      s[0] = wmma_bf16(qf[ks], kf0, s[0]);
      s[1] = wmma_bf16(qf[ks], kf1, s[1]);
    }

    // scale + causal mask in C-layout (m = v + 8*hi, n = lane%16)
    float pv[2][8];
#pragma unroll
    for (int nt = 0; nt < 2; ++nt)
#pragma unroll
      for (int v = 0; v < 8; ++v) {
        int m = q0 + v + 8 * hi;
        int t = t0 + nt * 16 + ln;
        float x = s[nt][v] * rscale;
        pv[nt][v] = (t <= m) ? x : -3.0e38f;
      }

    __syncthreads();  // ordering fence before re-filling Pt (1-wave block)
#pragma unroll
    for (int v = 0; v < 8; ++v) {
      float rmax = fmaxf(pv[0][v], pv[1][v]);
#pragma unroll
      for (int msk = 1; msk <= 8; msk <<= 1)
        rmax = fmaxf(rmax, __shfl_xor(rmax, msk, 32));
      float mnew = fmaxf(mprev[v], rmax);
      float p0 = __expf(pv[0][v] - mnew);
      float p1 = __expf(pv[1][v] - mnew);
      float pr = p0 + p1;
#pragma unroll
      for (int msk = 1; msk <= 8; msk <<= 1) pr += __shfl_xor(pr, msk, 32);
      float csc = __expf(mprev[v] - mnew);
      mprev[v] = mnew;
      lsum[v] = lsum[v] * csc + pr;
#pragma unroll
      for (int nb = 0; nb < 8; ++nb) o[nb][v] *= csc;
      Pt[(v + 8 * hi) * 32 + ln]      = f2bf(p0);
      Pt[(v + 8 * hi) * 32 + 16 + ln] = f2bf(p1);
    }
    __syncthreads();

    // O += P @ V : A = P (16x32, one WMMA K-step), B^T = Vt rows (contiguous)
    v16bf pf = load_frag(PtU, 0, 16);
#pragma unroll
    for (int nb = 0; nb < 8; ++nb) {
      v16bf vf = load_frag(Vh + (t0 >> 1), nb * 16, Sc / 2);
      o[nb] = wmma_bf16(pf, vf, o[nb]);
    }
  }

  // attn[(b*S + m), h*D + d] = O / l
#pragma unroll
  for (int nb = 0; nb < 8; ++nb)
#pragma unroll
    for (int v = 0; v < 8; ++v) {
      int m = q0 + v + 8 * hi;
      size_t row = (size_t)b * Sc + m;
      size_t col = (size_t)h * Dc + nb * 16 + ln;
      Ob[row * (size_t)HIDc + col] = f2bf(o[nb][v] / lsum[v]);
    }
}

}  // namespace

// ---------------------------------------------------------------------------
// Launch. Workspace layout (bytes, all 2-byte bf16; total 176 MB):
//   Xbf 32M | Wq/Wo(shared) 32M | Wk 8M | Wv 8M | Qp/attn(shared) 32M |
//   Kp 8M | Vp 8M | Qhead 32M | Khead 8M | Vt 8M
// ---------------------------------------------------------------------------
extern "C" void kernel_launch(void* const* d_in, const int* in_sizes, int n_in,
                              void* d_out, int out_size, void* d_ws, size_t ws_size,
                              hipStream_t stream) {
  (void)in_sizes; (void)n_in; (void)out_size; (void)ws_size;
  const float* X    = (const float*)d_in[0];
  const float* Wq   = (const float*)d_in[1];
  const float* Wk   = (const float*)d_in[2];
  const float* Wv   = (const float*)d_in[3];
  const float* Wo   = (const float*)d_in[4];
  const float* cosc = (const float*)d_in[5];
  const float* sinc = (const float*)d_in[6];
  const int*   pos  = (const int*)d_in[7];
  // d_in[8] = causal mask: implemented analytically in attn_kernel.

  char* ws = (char*)d_ws;
  size_t off = 0;
  auto take = [&](size_t bytes) { char* p = ws + off; off += bytes; return p; };
  uint16_t* Xbf = (uint16_t*)take((size_t)Mrows * HIDc * 2);
  uint16_t* Wqo = (uint16_t*)take((size_t)HIDc * HIDc * 2);         // Wq, then Wo
  uint16_t* Wkb = (uint16_t*)take((size_t)HIDc * NKVc * Dc * 2);
  uint16_t* Wvb = (uint16_t*)take((size_t)HIDc * NKVc * Dc * 2);
  uint16_t* QpA = (uint16_t*)take((size_t)Mrows * HIDc * 2);        // Qproj, then attn
  uint16_t* Kp  = (uint16_t*)take((size_t)Mrows * NKVc * Dc * 2);
  uint16_t* Vp  = (uint16_t*)take((size_t)Mrows * NKVc * Dc * 2);
  uint16_t* Qh  = (uint16_t*)take((size_t)Mrows * HIDc * 2);
  uint16_t* Kh  = (uint16_t*)take((size_t)Mrows * NKVc * Dc * 2);
  uint16_t* Vth = (uint16_t*)take((size_t)Mrows * NKVc * Dc * 2);

  auto cast = [&](const float* s, uint16_t* d, size_t n) {
    cast_f32_bf16<<<dim3((unsigned)((n + 511) / 512)), 512, 0, stream>>>(s, d, n);
  };
  cast(X,  Xbf, (size_t)Mrows * HIDc);
  cast(Wq, Wqo, (size_t)HIDc * HIDc);
  cast(Wk, Wkb, (size_t)HIDc * NKVc * Dc);
  cast(Wv, Wvb, (size_t)HIDc * NKVc * Dc);

  // Projections (bf16 out)
  gemm_bf16_kernel<true><<<dim3(HIDc / 128, Mrows / 128), 128, 0, stream>>>(
      Xbf, Wqo, QpA, HIDc, HIDc);
  gemm_bf16_kernel<true><<<dim3((NKVc * Dc) / 128, Mrows / 128), 128, 0, stream>>>(
      Xbf, Wkb, Kp, NKVc * Dc, HIDc);
  gemm_bf16_kernel<true><<<dim3((NKVc * Dc) / 128, Mrows / 128), 128, 0, stream>>>(
      Xbf, Wvb, Vp, NKVc * Dc, HIDc);

  // RoPE + relayout
  {
    size_t n = (size_t)Bc * Sc * NHc * (Dc / 2);
    rope_relayout<<<dim3((unsigned)((n + 255) / 256)), 256, 0, stream>>>(
        QpA, Qh, cosc, sinc, pos, NHc);
  }
  {
    size_t n = (size_t)Bc * Sc * NKVc * (Dc / 2);
    rope_relayout<<<dim3((unsigned)((n + 255) / 256)), 256, 0, stream>>>(
        Kp, Kh, cosc, sinc, pos, NKVc);
  }
  {
    size_t n = (size_t)Bc * Sc * NKVc * Dc;
    v_transpose<<<dim3((unsigned)((n + 255) / 256)), 256, 0, stream>>>(Vp, Vth);
  }

  // Flash attention (writes into QpA, whose Qproj contents are now dead)
  attn_kernel<<<dim3(Sc / 16, NHc, Bc), 32, 0, stream>>>(Qh, Kh, Vth, QpA);

  // Wo cast into Wqo (Wq contents dead after the Q projection)
  cast(Wo, Wqo, (size_t)HIDc * HIDc);

  // Output projection (fp32 out, straight to d_out)
  gemm_bf16_kernel<false><<<dim3(HIDc / 128, Mrows / 128), 128, 0, stream>>>(
      QpA, Wqo, d_out, HIDc, HIDc);
}
